// SNN_2654289789360
// MI455X (gfx1250) — matrix operand order repaired and checked by reading, not compiled, not run on hardware
//
#include <hip/hip_runtime.h>
#include <math.h>

typedef __attribute__((ext_vector_type(16))) _Float16 v16h;
typedef __attribute__((ext_vector_type(8)))  float    v8f;

#define T_STEPS   32
#define DDIM      256
#define B_TOTAL   131072
#define HALF_B    65536
#define HALF_FLAT 16777216u   // 2^24 : threefry counter half-offset for 2^25 elems

struct StepKeys { unsigned k0[T_STEPS]; unsigned k1[T_STEPS]; };

__host__ __device__ static inline unsigned rotl32(unsigned x, int r) {
  return (x << r) | (x >> (32 - r));
}

// Bit-exact JAX threefry2x32 (20 rounds, 5 key injections).
__host__ __device__ static inline void tf2x32(unsigned k0, unsigned k1,
                                              unsigned x0, unsigned x1,
                                              unsigned& o0, unsigned& o1) {
  const unsigned k2 = k0 ^ k1 ^ 0x1BD11BDAu;
  x0 += k0; x1 += k1;
#define TFR(r) { x0 += x1; x1 = rotl32(x1, (r)); x1 ^= x0; }
  TFR(13) TFR(15) TFR(26) TFR(6)
  x0 += k1; x1 += k2 + 1u;
  TFR(17) TFR(29) TFR(16) TFR(24)
  x0 += k2; x1 += k0 + 2u;
  TFR(13) TFR(15) TFR(26) TFR(6)
  x0 += k0; x1 += k1 + 3u;
  TFR(17) TFR(29) TFR(16) TFR(24)
  x0 += k1; x1 += k2 + 4u;
  TFR(13) TFR(15) TFR(26) TFR(6)
  x0 += k2; x1 += k0 + 5u;
#undef TFR
  o0 = x0; o1 = x1;
}

__device__ static inline float bits_to_uniform(unsigned b) {
  // JAX uniform: (bits>>9 | 0x3f800000) as float, minus 1 -> [0,1)
  return __uint_as_float((b >> 9) | 0x3f800000u) - 1.0f;
}

__global__ __launch_bounds__(256)
void snn_lif_wmma_kernel(const float* __restrict__ x,
                         const float* __restrict__ raw_w,
                         const float* __restrict__ readout_w,
                         const float* __restrict__ readout_b,
                         float* __restrict__ out,
                         StepKeys keys)
{
  const int lane = threadIdx.x & 31;
  const int wave = threadIdx.x >> 5;
  const int m  = lane & 15;   // A-matrix row within 16-row tile
  const int hl = lane >> 4;   // lane-half (selects K sub-block in frag layouts)
  const int n  = m;           // B/C-matrix column (H index; valid for n<4)

  // B fragments: w = sigmoid(raw_w), shape K x N = 32 x 16 per chunk,
// columns 4..15 zero-padded. 16-bit B layout: elem i of lane-half hl holds K=8*hl+(i&7)+16*(i>>3).
  __shared__ __align__(32) _Float16 bsh[8][32][16];
  if (wave == 0) {
    #pragma unroll 1
    for (int c = 0; c < 8; ++c) {
      #pragma unroll
      for (int i = 0; i < 16; ++i) {
        const int d = c * 32 + 8 * hl + (i & 7) + 16 * (i >> 3);
        float wv = 0.0f;
        if (n < 4) wv = 1.0f / (1.0f + expf(-raw_w[n * DDIM + d]));
        bsh[c][lane][i] = (_Float16)wv;
      }
    }
  }
  __syncthreads();

  const int tile = blockIdx.x * 8 + wave;       // 4096 tiles over lower half
  const int b0   = tile * 16;
  const unsigned jrow = (unsigned)(b0 + m) * DDIM;
  const float* xlo = x + (size_t)(b0 + m) * DDIM;            // lower-half row
  const float* xhi = x + (size_t)(b0 + HALF_B + m) * DDIM;   // paired upper row

  // Neuron state in WMMA C-layout: VGPR r -> row (r + 8*hl), lane%16 -> h.
  float ve_lo[8], th_lo[8], ss_lo[8];
  float ve_hi[8], th_hi[8], ss_hi[8];
  #pragma unroll
  for (int r = 0; r < 8; ++r) {
    ve_lo[r] = 0.0f; th_lo[r] = 1.0f; ss_lo[r] = 0.0f;
    ve_hi[r] = 0.0f; th_hi[r] = 1.0f; ss_hi[r] = 0.0f;
  }

  #pragma unroll 1
  for (int t = 0; t < T_STEPS; ++t) {
    const unsigned K0 = keys.k0[t];
    const unsigned K1 = keys.k1[t];
    v8f acc_lo = {};
    v8f acc_hi = {};
    #pragma unroll 1
    for (int c = 0; c < 8; ++c) {
      v16h a_lo = {};
      v16h a_hi = {};
      // 16-bit A layout: frag elem (8*g+i) -> K = 8*hl + i + 16*g within chunk
      #pragma unroll
      for (int g = 0; g < 2; ++g) {
        const int dbase = c * 32 + 16 * g + 8 * hl;
        const float4 l0 = *(const float4*)(xlo + dbase);
        const float4 l1 = *(const float4*)(xlo + dbase + 4);
        const float4 h0 = *(const float4*)(xhi + dbase);
        const float4 h1 = *(const float4*)(xhi + dbase + 4);
        float xl[8], xh[8];
        xl[0]=l0.x; xl[1]=l0.y; xl[2]=l0.z; xl[3]=l0.w;
        xl[4]=l1.x; xl[5]=l1.y; xl[6]=l1.z; xl[7]=l1.w;
        xh[0]=h0.x; xh[1]=h0.y; xh[2]=h0.z; xh[3]=h0.w;
        xh[4]=h1.x; xh[5]=h1.y; xh[6]=h1.z; xh[7]=h1.w;
        #pragma unroll
        for (int i = 0; i < 8; ++i) {
          const unsigned j = jrow + (unsigned)(dbase + i);
          unsigned o0, o1;
          tf2x32(K0, K1, j, j + HALF_FLAT, o0, o1);  // o0 -> row b, o1 -> row b+65536
          const float u0 = bits_to_uniform(o0);
          const float u1 = bits_to_uniform(o1);
          a_lo[8 * g + i] = (u0 < xl[i]) ? (_Float16)1.0f : (_Float16)0.0f;
          a_hi[8 * g + i] = (u1 < xh[i]) ? (_Float16)1.0f : (_Float16)0.0f;
        }
      }
      const v16h bf = *(const v16h*)&bsh[c][lane][0];
      acc_lo = __builtin_amdgcn_wmma_f32_16x16x32_f16(
          false, a_lo, false, bf, (short)0, acc_lo, false, false);
      acc_hi = __builtin_amdgcn_wmma_f32_16x16x32_f16(
          false, a_hi, false, bf, (short)0, acc_hi, false, false);
    }

    // LIF dynamics (note: v_i == 0 invariant => s_i == s_e)
    const float IES = 12.0f / 256.0f;
    float se_lo[8], se_hi[8];
    #pragma unroll
    for (int r = 0; r < 8; ++r) {
      float ie = acc_lo[r] * IES;
      float v  = ve_lo[r] + (ie - ve_lo[r]) * 0.5f;
      float se = (v >= th_lo[r]) ? 1.0f : 0.0f;
      v *= (1.0f - se);
      th_lo[r] = th_lo[r] + (1.0f - th_lo[r]) * 0.02f + 0.05f * se;
      ve_lo[r] = v; se_lo[r] = se; ss_lo[r] += se;

      ie = acc_hi[r] * IES;
      v  = ve_hi[r] + (ie - ve_hi[r]) * 0.5f;
      se = (v >= th_hi[r]) ? 1.0f : 0.0f;
      v *= (1.0f - se);
      th_hi[r] = th_hi[r] + (1.0f - th_hi[r]) * 0.02f + 0.05f * se;
      ve_hi[r] = v; se_hi[r] = se; ss_hi[r] += se;
    }
    // lateral inhibition: v_e -= (sum_h s_e) - s_e ; sum over 4-lane N-group
    #pragma unroll
    for (int r = 0; r < 8; ++r) {
      float a = se_lo[r] + __shfl_xor(se_lo[r], 1);
      a += __shfl_xor(a, 2);
      ve_lo[r] -= (a - se_lo[r]);
      float b = se_hi[r] + __shfl_xor(se_hi[r], 1);
      b += __shfl_xor(b, 2);
      ve_hi[r] -= (b - se_hi[r]);
    }
  }

  // readout: out[b] = sum_h (ss/32) * rw[h] + bias
  const float rw   = (n < 4) ? readout_w[n] : 0.0f;
  const float bias = readout_b[0];
  const float inv_t = 1.0f / (float)T_STEPS;
  #pragma unroll
  for (int r = 0; r < 8; ++r) {
    float v = ss_lo[r] * inv_t * rw;
    v += __shfl_xor(v, 1);
    v += __shfl_xor(v, 2);
    if (n == 0) out[b0 + 8 * hl + r] = v + bias;
    float v2 = ss_hi[r] * inv_t * rw;
    v2 += __shfl_xor(v2, 1);
    v2 += __shfl_xor(v2, 2);
    if (n == 0) out[b0 + HALF_B + 8 * hl + r] = v2 + bias;
  }
}

extern "C" void kernel_launch(void* const* d_in, const int* in_sizes, int n_in,
                              void* d_out, int out_size, void* d_ws, size_t ws_size,
                              hipStream_t stream) {
  (void)in_sizes; (void)n_in; (void)out_size; (void)d_ws; (void)ws_size;
  const float* x    = (const float*)d_in[0];
  const float* rw   = (const float*)d_in[1];
  const float* rowt = (const float*)d_in[2];
  const float* rob  = (const float*)d_in[3];
  float* out = (float*)d_out;

  // keys = jax.random.split(jax.random.key(42), 32) computed host-side,
  // bit-exact: split -> random_bits shape (32,2): counts iota(64), halves.
  StepKeys sk;
  unsigned bits[2 * T_STEPS];
  for (unsigned i = 0; i < T_STEPS; ++i) {
    unsigned o0, o1;
    tf2x32(0u, 42u, i, i + T_STEPS, o0, o1);
    bits[i] = o0; bits[i + T_STEPS] = o1;
  }
  for (int t = 0; t < T_STEPS; ++t) {
    sk.k0[t] = bits[2 * t];
    sk.k1[t] = bits[2 * t + 1];
  }

  dim3 grid(HALF_B / 16 / 8);  // 512 blocks x 8 waves = 4096 paired tiles
  dim3 block(256);
  snn_lif_wmma_kernel<<<grid, block, 0, stream>>>(x, rw, rowt, rob, out, sk);
}
